// SwinTransformerBlock_72481868087425
// MI455X (gfx1250) — matrix operand order repaired and checked by reading, not compiled, not run on hardware
//
#include <hip/hip_runtime.h>
#include <hip/hip_bf16.h>
#include <math.h>

typedef __bf16 bf16_t;
typedef __attribute__((ext_vector_type(16))) __bf16 v16bf;
typedef __attribute__((ext_vector_type(8)))  float  v8f;

#define WIN       7
#define N_TOK     49
#define HEADS     8
#define HEAD_DIM  32
#define C_DIM     256
#define H_IMG     56
#define SHIFT_SZ  3
#define M_TOT     200704   // 64 batches * 64 windows * 49 tokens
#define NWIN_TOT  4096     // 64 * 64

// ---------------------------------------------------------------------------
// WMMA fragment helpers (layouts per CDNA5 ISA 7.12.2, wave32)
// ---------------------------------------------------------------------------
// A (16x32 bf16): lanes 0-15 row M=lane hold K {0..7,16..23}; lanes 16-31 hold
// K {8..15,24..31}.  base points at [row0][k0] of a row-major LDS tile.
__device__ __forceinline__ v16bf load_a_frag(const bf16_t* base, int lda) {
  const int lane = threadIdx.x & 31;
  const bf16_t* p = base + (lane & 15) * lda + ((lane >> 4) << 3);
  union { v16bf v; bf16_t e[16]; } u;
#pragma unroll
  for (int i = 0; i < 8; ++i) u.e[i] = p[i];
#pragma unroll
  for (int i = 0; i < 8; ++i) u.e[8 + i] = p[16 + i];
  return u.v;
}

// B (32x16 bf16): lane n<16 = column n, K {0..7,16..23}; lane n+16 = column n, K {8..15,24..31}.
// LDS tile is stored K-transposed: Bs[n][k], so each lane reads two runs of 8 contiguous k.
__device__ __forceinline__ v16bf load_b_frag(const bf16_t* base, int ldb) {
  const int lane = threadIdx.x & 31;
  const bf16_t* p = base + (lane & 15) * ldb + ((lane >> 4) << 3);
  union { v16bf v; bf16_t e[16]; } u;
#pragma unroll
  for (int i = 0; i < 16; ++i) u.e[i] = p[i + (i & 8)];
  return u.v;
}

__device__ __forceinline__ v8f wmma_bf16f32(v16bf a, v16bf b, v8f c) {
  // (neg_a, A, neg_b, B, c_mod, C, reuse_a, reuse_b)
  return __builtin_amdgcn_wmma_f32_16x16x32_bf16(false, a, false, b, (short)0, c,
                                                 false, false);
}

// ---------------------------------------------------------------------------
// fp32 -> bf16 weight cast
// ---------------------------------------------------------------------------
__global__ __launch_bounds__(256) void cast_kernel(const float* __restrict__ a,
                                                   bf16_t* __restrict__ o, int n) {
  int i = blockIdx.x * 256 + threadIdx.x;
  if (i < n) o[i] = (bf16_t)a[i];
}

// ---------------------------------------------------------------------------
// LayerNorm (+ optional shift/window-partition gather), fp32 in -> bf16 out
// one token per block, 256 threads = 256 channels
// ---------------------------------------------------------------------------
__global__ __launch_bounds__(256) void ln_kernel(const float* __restrict__ x,
                                                 const float* __restrict__ g,
                                                 const float* __restrict__ bv,
                                                 bf16_t* __restrict__ out,
                                                 int shifted) {
  __shared__ float red[256];
  const int r = blockIdx.x;
  const int c = threadIdx.x;
  size_t src;
  if (shifted) {
    // output row r is in window order; gather from rolled image position
    int w  = r / N_TOK, tt = r - w * N_TOK;
    int b  = w >> 6, wi = w & 63;
    int wh = wi >> 3, ww = wi & 7;
    int i  = tt / WIN, j = tt - i * WIN;
    int hh = wh * WIN + i + SHIFT_SZ; if (hh >= H_IMG) hh -= H_IMG;
    int wc = ww * WIN + j + SHIFT_SZ; if (wc >= H_IMG) wc -= H_IMG;
    src = ((size_t)b * (H_IMG * H_IMG) + hh * H_IMG + wc) * C_DIM;
  } else {
    src = (size_t)r * C_DIM;
  }
  float v = x[src + c];
  red[c] = v; __syncthreads();
#pragma unroll
  for (int s = 128; s > 0; s >>= 1) { if (c < s) red[c] += red[c + s]; __syncthreads(); }
  float mu = red[0] * (1.0f / 256.0f);
  __syncthreads();
  float d = v - mu;
  red[c] = d * d; __syncthreads();
#pragma unroll
  for (int s = 128; s > 0; s >>= 1) { if (c < s) red[c] += red[c + s]; __syncthreads(); }
  float var = red[0] * (1.0f / 256.0f);
  float y = d * rsqrtf(var + 1e-5f) * g[c] + bv[c];
  out[(size_t)r * C_DIM + c] = (bf16_t)y;
}

// ---------------------------------------------------------------------------
// bf16 WMMA GEMM: C[M,N] = A[M,K] * B[K,N]
// block tile 128x128, 8 waves (256 thr), BK=32; each wave owns 32x64 sub-tile
// -> 8 WMMAs per wave per K-step (2 A-frags x 4 B-frags)
// EPI 0: +bias, scatter q(*scale)/k/v     EPI 1: +bias +resid, window-reverse
// EPI 2: +bias, gelu -> bf16              EPI 3: +bias +resid -> f32 out
// ---------------------------------------------------------------------------
template <int EPI>
__global__ __launch_bounds__(256)
void gemm_kernel(const bf16_t* __restrict__ A, const bf16_t* __restrict__ Bm,
                 int K, int N, const float* __restrict__ bias,
                 const float* __restrict__ resid, float* __restrict__ out_f32,
                 bf16_t* __restrict__ oq, bf16_t* __restrict__ ok,
                 bf16_t* __restrict__ ov, bf16_t* __restrict__ out_bf) {
  __shared__ __align__(16) bf16_t As[128][40];  // row-major [m][k]
  __shared__ __align__(16) bf16_t Bs[128][40];  // transposed [n][k]

  const int m0   = blockIdx.y * 128;
  const int n0   = blockIdx.x * 128;
  const int t    = threadIdx.x;
  const int wave = t >> 5;
  const int lane = t & 31;
  const int wm   = wave & 3;   // row block: 32 rows
  const int wn   = wave >> 2;  // col block: 64 cols

  v8f zero = {};
  v8f acc[2][4];
#pragma unroll
  for (int mi = 0; mi < 2; ++mi)
#pragma unroll
    for (int nt = 0; nt < 4; ++nt) acc[mi][nt] = zero;

  for (int k0 = 0; k0 < K; k0 += 32) {
    // stage A tile 128x32 (16B vector loads, 2 passes)
#pragma unroll
    for (int p = 0; p < 2; ++p) {
      int e = p * 2048 + t * 8;
      int r = e >> 5, c = e & 31;
      const bf16_t* gp = A + (size_t)(m0 + r) * K + (k0 + c);
      *(uint4*)&As[r][c] = *(const uint4*)gp;
    }
    // stage B tile 32x128, transposed into Bs[n][k]
#pragma unroll
    for (int p = 0; p < 2; ++p) {
      int e  = p * 2048 + t * 8;
      int kk = e >> 7, n = e & 127;
      union { uint4 u; bf16_t e8[8]; } tmp;
      tmp.u = *(const uint4*)(Bm + (size_t)(k0 + kk) * N + (n0 + n));
#pragma unroll
      for (int j = 0; j < 8; ++j) Bs[n + j][kk] = tmp.e8[j];
    }
    if (k0 + 32 < K) {
      __builtin_prefetch(A + (size_t)(m0 + (t & 127)) * K + (k0 + 32), 0, 1);
      __builtin_prefetch(Bm + (size_t)(k0 + 32 + (t & 31)) * N + n0, 0, 1);
    }
    __syncthreads();

    v16bf af[2];
#pragma unroll
    for (int mi = 0; mi < 2; ++mi)
      af[mi] = load_a_frag(&As[wm * 32 + mi * 16][0], 40);
#pragma unroll
    for (int nt = 0; nt < 4; ++nt) {
      v16bf bfrag = load_b_frag(&Bs[wn * 64 + nt * 16][0], 40);
#pragma unroll
      for (int mi = 0; mi < 2; ++mi)
        acc[mi][nt] = wmma_bf16f32(af[mi], bfrag, acc[mi][nt]);
    }
    __syncthreads();
  }

  // epilogue: lane holds C[.. + 8*(lane>>4)+j][.. + (lane&15)]
  const int mb = (lane >> 4) << 3;
  const int nc = lane & 15;
#pragma unroll
  for (int mi = 0; mi < 2; ++mi) {
#pragma unroll
    for (int nt = 0; nt < 4; ++nt) {
#pragma unroll
      for (int j = 0; j < 8; ++j) {
        float val = acc[mi][nt][j];
        int m = m0 + wm * 32 + mi * 16 + mb + j;
        int n = n0 + wn * 64 + nt * 16 + nc;
        val += bias[n];
        if constexpr (EPI == 0) {           // QKV scatter
          int s = n >> 8, rem = n & 255;
          int hh = rem >> 5, d = rem & 31;
          int w = m / N_TOK, tt = m - w * N_TOK;
          size_t dst = ((size_t)(w * HEADS + hh) * N_TOK + tt) * HEAD_DIM + d;
          if (s == 0)      oq[dst] = (bf16_t)(val * 0.17677669529663687f);
          else if (s == 1) ok[dst] = (bf16_t)val;
          else             ov[dst] = (bf16_t)val;
        } else if constexpr (EPI == 1) {    // proj: window-reverse + unshift + resid
          int w  = m / N_TOK, tt = m - w * N_TOK;
          int b  = w >> 6, wi = w & 63;
          int wh = wi >> 3, ww = wi & 7;
          int i  = tt / WIN, jj = tt - i * WIN;
          int hh = wh * WIN + i + SHIFT_SZ; if (hh >= H_IMG) hh -= H_IMG;
          int wc = ww * WIN + jj + SHIFT_SZ; if (wc >= H_IMG) wc -= H_IMG;
          size_t idx = ((size_t)b * (H_IMG * H_IMG) + hh * H_IMG + wc) * C_DIM + n;
          out_f32[idx] = resid[idx] + val;
        } else if constexpr (EPI == 2) {    // mlp1: exact gelu -> bf16
          val = 0.5f * val * (1.0f + erff(val * 0.70710678118654752f));
          out_bf[(size_t)m * N + n] = (bf16_t)val;
        } else {                            // mlp2: + resid -> out
          size_t idx = (size_t)m * C_DIM + n;
          out_f32[idx] = resid[idx] + val;
        }
      }
    }
  }
}

// ---------------------------------------------------------------------------
// Windowed attention: one block per (window, head), 128 threads (4 waves)
// scores = Q@K^T (WMMA) + rel-pos bias + shift mask; softmax; O = P@V (WMMA)
// ---------------------------------------------------------------------------
__global__ __launch_bounds__(128)
void attn_kernel(const bf16_t* __restrict__ q, const bf16_t* __restrict__ k,
                 const bf16_t* __restrict__ v, const float* __restrict__ rpb,
                 bf16_t* __restrict__ o) {
  __shared__ __align__(16) bf16_t Qs[64][40];   // [query][dim]   (A for scores)
  __shared__ __align__(16) bf16_t Ks[64][40];   // [key][dim]     (B^T for scores)
  __shared__ __align__(16) bf16_t Vt[32][72];   // [dim][key]     (B^T for output)
  __shared__ __align__(16) float  S[64][64];    // fp32 scores
  __shared__ __align__(16) bf16_t P[64][72];    // bf16 probs     (A for output)

  const int wh_id = blockIdx.x;           // window*8 + head
  const int w     = wh_id >> 3;
  const int head  = wh_id & 7;
  const int t     = threadIdx.x;
  const int wave  = t >> 5;
  const int lane  = t & 31;
  const size_t base = (size_t)wh_id * N_TOK * HEAD_DIM;

  for (int e = t; e < 64 * 32; e += 128) {
    int r = e >> 5, c = e & 31;
    bf16_t qv = (bf16_t)0.0f, kv = (bf16_t)0.0f, vv = (bf16_t)0.0f;
    if (r < N_TOK) {
      qv = q[base + r * 32 + c];
      kv = k[base + r * 32 + c];
      vv = v[base + r * 32 + c];
    }
    Qs[r][c] = qv; Ks[r][c] = kv; Vt[c][r] = vv;
  }
  __syncthreads();

  // ---- scores: each wave owns 16 query rows, K=32 in one WMMA
  {
    const int wi  = w & 63;
    const int wwh = wi >> 3, www = wi & 7;
    v16bf afrag = load_a_frag(&Qs[wave * 16][0], 40);
#pragma unroll
    for (int nt = 0; nt < 4; ++nt) {
      v8f acc = {};
      v16bf bfrag = load_b_frag(&Ks[nt * 16][0], 40);
      acc = wmma_bf16f32(afrag, bfrag, acc);
      int n = nt * 16 + (lane & 15);
#pragma unroll
      for (int j = 0; j < 8; ++j) {
        int m = wave * 16 + ((lane >> 4) << 3) + j;
        float val = acc[j];
        if (m < N_TOK && n < N_TOK) {
          int qi = m / WIN, qj = m - qi * WIN;
          int ki = n / WIN, kj = n - ki * WIN;
          val += rpb[((qi - ki + 6) * 13 + (qj - kj + 6)) * HEADS + head];
          int hq = wwh * WIN + qi, wq = www * WIN + qj;
          int hk = wwh * WIN + ki, wk = www * WIN + kj;
          int rq = (hq < 49 ? 0 : (hq < 53 ? 1 : 2)) * 3 + (wq < 49 ? 0 : (wq < 53 ? 1 : 2));
          int rk = (hk < 49 ? 0 : (hk < 53 ? 1 : 2)) * 3 + (wk < 49 ? 0 : (wk < 53 ? 1 : 2));
          if (rq != rk) val -= 100.0f;
        } else {
          val = -1e30f;
        }
        S[m][n] = val;
      }
    }
  }
  __syncthreads();

  // ---- softmax per query row (fp32), write bf16 probs with zero padding
  if (t < 64) {
    const int m = t;
    if (m < N_TOK) {
      float mx = -1e30f;
      for (int n2 = 0; n2 < N_TOK; ++n2) mx = fmaxf(mx, S[m][n2]);
      float sum = 0.0f;
      for (int n2 = 0; n2 < N_TOK; ++n2) {
        float e2 = __expf(S[m][n2] - mx);
        S[m][n2] = e2;
        sum += e2;
      }
      float inv = 1.0f / sum;
      for (int n2 = 0; n2 < 64; ++n2)
        P[m][n2] = (bf16_t)((n2 < N_TOK) ? S[m][n2] * inv : 0.0f);
    } else {
      for (int n2 = 0; n2 < 64; ++n2) P[m][n2] = (bf16_t)0.0f;
    }
  }
  __syncthreads();

  // ---- O = P(64x64) @ V(64x32): 2 K-steps, 2 N-tiles per wave
  v8f zero = {};
  v8f oacc[2] = {zero, zero};
#pragma unroll
  for (int kk = 0; kk < 2; ++kk) {
    v16bf afrag = load_a_frag(&P[wave * 16][kk * 32], 72);
#pragma unroll
    for (int nt = 0; nt < 2; ++nt) {
      v16bf bfrag = load_b_frag(&Vt[nt * 16][kk * 32], 72);
      oacc[nt] = wmma_bf16f32(afrag, bfrag, oacc[nt]);
    }
  }
#pragma unroll
  for (int nt = 0; nt < 2; ++nt) {
#pragma unroll
    for (int j = 0; j < 8; ++j) {
      int m = wave * 16 + ((lane >> 4) << 3) + j;
      int d = nt * 16 + (lane & 15);
      if (m < N_TOK)
        o[((size_t)w * N_TOK + m) * C_DIM + head * HEAD_DIM + d] = (bf16_t)oacc[nt][j];
    }
  }
}

// ---------------------------------------------------------------------------
extern "C" void kernel_launch(void* const* d_in, const int* in_sizes, int n_in,
                              void* d_out, int out_size, void* d_ws, size_t ws_size,
                              hipStream_t stream) {
  (void)in_sizes; (void)n_in; (void)out_size; (void)ws_size;
  const float* x      = (const float*)d_in[0];
  const float* ln1_g  = (const float*)d_in[1];
  const float* ln1_b  = (const float*)d_in[2];
  const float* qkv_w  = (const float*)d_in[3];
  const float* qkv_b  = (const float*)d_in[4];
  const float* rpb    = (const float*)d_in[5];
  const float* proj_w = (const float*)d_in[6];
  const float* proj_b = (const float*)d_in[7];
  const float* ln2_g  = (const float*)d_in[8];
  const float* ln2_b  = (const float*)d_in[9];
  const float* mlp_w1 = (const float*)d_in[10];
  const float* mlp_b1 = (const float*)d_in[11];
  const float* mlp_w2 = (const float*)d_in[12];
  const float* mlp_b2 = (const float*)d_in[13];
  float* out = (float*)d_out;

  char* ws = (char*)d_ws;
  size_t off = 0;
  auto alloc = [&](size_t bytes) -> char* {
    char* p = ws + off;
    off += (bytes + 255) & ~(size_t)255;
    return p;
  };
  bf16_t* qkvw_bf  = (bf16_t*)alloc((size_t)256 * 768 * 2);
  bf16_t* projw_bf = (bf16_t*)alloc((size_t)256 * 256 * 2);
  bf16_t* w1_bf    = (bf16_t*)alloc((size_t)256 * 1024 * 2);
  bf16_t* w2_bf    = (bf16_t*)alloc((size_t)1024 * 256 * 2);
  float*  x1       = (float*)alloc((size_t)M_TOT * 256 * 4);   // post-attn residual
  bf16_t* bufA     = (bf16_t*)alloc((size_t)M_TOT * 256 * 2);  // y -> o -> h
  bf16_t* big      = (bf16_t*)alloc((size_t)M_TOT * 1024 * 2); // qkv -> gelu acts
  bf16_t* qb = big;
  bf16_t* kb = big + (size_t)M_TOT * 256;
  bf16_t* vb = big + (size_t)M_TOT * 512;
  bf16_t* gb = big;

  // weights -> bf16
  cast_kernel<<<(256 * 768 + 255) / 256, 256, 0, stream>>>(qkv_w, qkvw_bf, 256 * 768);
  cast_kernel<<<(256 * 256 + 255) / 256, 256, 0, stream>>>(proj_w, projw_bf, 256 * 256);
  cast_kernel<<<(256 * 1024 + 255) / 256, 256, 0, stream>>>(mlp_w1, w1_bf, 256 * 1024);
  cast_kernel<<<(1024 * 256 + 255) / 256, 256, 0, stream>>>(mlp_w2, w2_bf, 1024 * 256);

  // LN1 + roll + window partition
  ln_kernel<<<M_TOT, 256, 0, stream>>>(x, ln1_g, ln1_b, bufA, 1);

  // QKV GEMM (M=200704, N=768, K=256)
  gemm_kernel<0><<<dim3(768 / 128, M_TOT / 128), 256, 0, stream>>>(
      bufA, qkvw_bf, 256, 768, qkv_b, nullptr, nullptr, qb, kb, vb, nullptr);

  // windowed attention (4096 windows * 8 heads), writes o into bufA
  attn_kernel<<<NWIN_TOT * HEADS, 128, 0, stream>>>(qb, kb, vb, rpb, bufA);

  // proj GEMM + window reverse + unshift + residual -> x1 (f32, image order)
  gemm_kernel<1><<<dim3(256 / 128, M_TOT / 128), 256, 0, stream>>>(
      bufA, projw_bf, 256, 256, proj_b, x, x1, nullptr, nullptr, nullptr, nullptr);

  // LN2 -> bf16 (image order)
  ln_kernel<<<M_TOT, 256, 0, stream>>>(x1, ln2_g, ln2_b, bufA, 0);

  // MLP1 GEMM + exact GELU (M=200704, N=1024, K=256)
  gemm_kernel<2><<<dim3(1024 / 128, M_TOT / 128), 256, 0, stream>>>(
      bufA, w1_bf, 256, 1024, mlp_b1, nullptr, nullptr, nullptr, nullptr, nullptr, gb);

  // MLP2 GEMM + residual -> out (M=200704, N=256, K=1024)
  gemm_kernel<3><<<dim3(256 / 128, M_TOT / 128), 256, 0, stream>>>(
      gb, w2_bf, 1024, 256, mlp_b2, x1, out, nullptr, nullptr, nullptr, nullptr);
}
